// RAMGen_12077448037141
// MI455X (gfx1250) — compile-verified, matrix operand order checked by reading, not run on hardware
//
#include <hip/hip_runtime.h>
#include <cmath>

// ---------------- problem constants ----------------
constexpr int N_    = 64;
constexpr int M_    = 2;
constexpr int CIN_  = 3;
constexpr int T_    = 300;
constexpr int V_    = 25;
constexpr int VP_   = 32;   // V padded to 2 WMMA tiles
constexpr int K_    = 3;
constexpr int CENC_ = 64;
constexpr int O_    = 64;
constexpr int INTER_= 16;
constexpr float EPS_BN = 1e-5f;

typedef __attribute__((ext_vector_type(16))) _Float16 v16h;
typedef __attribute__((ext_vector_type(8)))  float    v8f;
typedef unsigned int v4u_ __attribute__((ext_vector_type(4)));
typedef int          v8i_ __attribute__((ext_vector_type(8)));
typedef int          v4i_ __attribute__((ext_vector_type(4)));

#ifndef __has_builtin
#define __has_builtin(x) 0
#endif
#if __has_builtin(__builtin_amdgcn_tensor_load_to_lds) && \
    !__has_include(<hip/amd_detail/amd_gfx1250_TDM.h>)
#define USE_TDM 1
#else
#define USE_TDM 0
#endif

// Fragment-ready LDS helpers (ISA 7.12.2, 16-bit A 16x32 / B 32x16):
//   operand lane l: m = l&15 (row/col), h = l>>4 (K-half)
//   element e:      K = (e&7) + 8*h + 16*(e>>3)
// Inverse (producer side), given K index kk and m:
//   h = (kk>>3)&1, e = (kk&7) + 8*(kk>>4), lane = (h<<4)|m
__device__ __forceinline__ int frag_lane(int kk, int m) {
  return (((kk >> 3) & 1) << 4) | m;
}
__device__ __forceinline__ int frag_elem(int kk) {
  return (kk & 7) + 8 * (kk >> 4);
}

// ============================================================
// K_pre: split x (N,M,C,T,V) f32 into xa,xb f16 (N,C,T,V)
// ============================================================
__global__ void split_kernel(const float* __restrict__ x,
                             _Float16* __restrict__ xa,
                             _Float16* __restrict__ xb) {
  const int per = CIN_ * T_ * V_;
  const int total = N_ * per;
  for (int idx = blockIdx.x * blockDim.x + threadIdx.x; idx < total;
       idx += gridDim.x * blockDim.x) {
    int n = idx / per, r = idx % per;
    xa[idx] = (_Float16)x[(n * M_ + 0) * per + r];
    xb[idx] = (_Float16)x[(n * M_ + 1) * per + r];
  }
}

// ============================================================
// K0: RAM_g (N,T,V,V) and ab mask (N,T) from raw f32 x
// ============================================================
__global__ void ramg_kernel(const float* __restrict__ x,
                            float* __restrict__ ramg,
                            float* __restrict__ ab) {
  int nt = blockIdx.x;            // N_*T_ blocks, 128 threads
  int n = nt / T_, t = nt % T_;
  __shared__ float xaL[CIN_][V_], xbL[CIN_][V_];
  int tid = threadIdx.x;
  if (tid < CIN_ * V_) {
    int c = tid / V_, v = tid % V_;
    xaL[c][v] = x[(((n * M_ + 0) * CIN_ + c) * T_ + t) * V_ + v];
    xbL[c][v] = x[(((n * M_ + 1) * CIN_ + c) * T_ + t) * V_ + v];
  }
  __syncthreads();
  if (tid == 0) {
    float tot = 0.f;
    for (int c = 0; c < CIN_; ++c) {
      float m = 0.f;
      for (int v = 0; v < V_; ++v) m += xbL[c][v];
      m /= V_;
      float ss = 0.f;
      for (int v = 0; v < V_; ++v) { float d = xbL[c][v] - m; ss += d * d; }
      tot += ss / (V_ - 1);
    }
    ab[nt] = (tot < 1e-4f) ? 0.f : 1.f;
  }
  for (int idx = tid; idx < V_ * V_; idx += 128) {
    int v = idx / V_, w = idx % V_;
    float s = 0.f;
    for (int c = 0; c < CIN_; ++c) {
      float lg = logf(expf(xaL[c][v] - xbL[c][w]) + 1e-5f);
      s += lg * lg;
    }
    ramg[(size_t)nt * V_ * V_ + idx] = expf(-s * (1.f / 3.f));
  }
}

// ============================================================
// K1: att_raw[n,k,v,w] = (1/(16*300)) * sum_{t,i} fa * fb
// fa/fb produced on the fly into fragment-ready LDS; each wave
// fetches its v16h operand as one contiguous 32B LDS load.
// One block per (n,k); 4 waves = 2x2 tiling of padded 32x32.
// ============================================================
__global__ __launch_bounds__(128)
void att_kernel(const _Float16* __restrict__ xh,   // (N,C,T,V)
                const float* __restrict__ wa, const float* __restrict__ ba,
                const float* __restrict__ wb, const float* __restrict__ bb,
                int C, float* __restrict__ att /* (N,K,V,V) */) {
  int nk = blockIdx.x;
  int n = nk / K_, k = nk % K_;
  __shared__ __align__(32) _Float16 faF[2][32][16];  // [v-tile][lane][elem]
  __shared__ __align__(32) _Float16 fbF[2][32][16];  // [w-tile][lane][elem]
  __shared__ float   waL[INTER_ * CENC_], wbL[INTER_ * CENC_];
  __shared__ float   baL[INTER_], bbL[INTER_];
  __shared__ _Float16 xL[CENC_][2 * V_];
  int tid = threadIdx.x;

  for (int idx = tid; idx < INTER_ * C; idx += 128) {
    int i = idx / C, c = idx % C;
    waL[idx] = wa[(k * INTER_ + i) * C + c];
    wbL[idx] = wb[(k * INTER_ + i) * C + c];
  }
  if (tid < INTER_) { baL[tid] = ba[k * INTER_ + tid]; bbL[tid] = bb[k * INTER_ + tid]; }
  __syncthreads();

  int lane = tid & 31, wv = tid >> 5;
  int v0 = (wv & 1) * 16, w0 = (wv >> 1) * 16;
  int h = lane >> 4, m = lane & 15;
  v8f acc = {};

  for (int t0 = 0; t0 < T_; t0 += 2) {       // 150 chunks of K=32 (2 t's x 16 i's)
    for (int idx = tid; idx < C * 2 * V_; idx += 128) {
      int c = idx / (2 * V_), rem = idx % (2 * V_);
      int dt = rem / V_, v = rem % V_;
      xL[c][dt * V_ + v] = xh[((size_t)(n * C + c) * T_ + (t0 + dt)) * V_ + v];
    }
    __syncthreads();
    for (int idx = tid; idx < 32 * VP_; idx += 128) {
      int kk = idx >> 5, v = idx & 31;     // kk = dt*16 + i (contraction index)
      int dt = kk >> 4, i = kk & 15;
      float fa = 0.f, fb = 0.f;
      if (v < V_) {
        fa = baL[i]; fb = bbL[i];
        for (int c = 0; c < C; ++c) {
          float xv = (float)xL[c][dt * V_ + v];
          fa += waL[i * C + c] * xv;
          fb += wbL[i * C + c] * xv;
        }
      }
      int f = v >> 4, mm = v & 15;
      int ln = frag_lane(kk, mm), e = frag_elem(kk);
      faF[f][ln][e] = (_Float16)fa;
      fbF[f][ln][e] = (_Float16)fb;
    }
    __syncthreads();
    v16h a = *reinterpret_cast<const v16h*>(&faF[wv & 1][lane][0]);
    v16h b = *reinterpret_cast<const v16h*>(&fbF[wv >> 1][lane][0]);
    acc = __builtin_amdgcn_wmma_f32_16x16x32_f16(false, a, false, b, (short)0,
                                                 acc, false, false);
    __syncthreads();
  }

  const float scale = 1.f / (float)(INTER_ * T_);
#pragma unroll
  for (int j = 0; j < 8; ++j) {            // C/D layout: row = j + 8*h, col = lane&15
    int vv = v0 + j + 8 * h;
    int ww = w0 + m;
    if (vv < V_ && ww < V_)
      att[((size_t)(n * K_ + k) * V_ + vv) * V_ + ww] = acc[j] * scale;
  }
}

// ============================================================
// K2: in-place softmax over v (axis -2) per (n,k,w) column + Ahat
// ============================================================
__global__ void softmax_kernel(float* __restrict__ att,
                               const float* __restrict__ A,
                               const float* __restrict__ PA) {
  int nk = blockIdx.x;
  int k = nk % K_;
  int w = threadIdx.x;
  if (w >= V_) return;
  float col[V_];
  float mx = -3.4e38f;
  for (int v = 0; v < V_; ++v) {
    col[v] = att[((size_t)nk * V_ + v) * V_ + w];
    mx = fmaxf(mx, col[v]);
  }
  float s = 0.f;
  for (int v = 0; v < V_; ++v) { col[v] = expf(col[v] - mx); s += col[v]; }
  float inv = 1.f / s;
  for (int v = 0; v < V_; ++v) {
    float ah = A[(k * V_ + v) * V_ + w] + PA[(k * V_ + v) * V_ + w];
    att[((size_t)nk * V_ + v) * V_ + w] = col[v] * inv + ah;
  }
}

// ============================================================
// K3: y[n,o,t,w] = bdsum[o] + sum_{k,c} wd[k,o,c] * (sum_v x[n,c,t,v]*att[n,k,v,w])
// Block = (n,w). GEMM: M = t-chunk(32), N = 64, K = K_*C (pad 32, up to 192).
// Z and WD staged in fragment-ready LDS; v16h operands load as 32B.
// ============================================================
__global__ __launch_bounds__(128)
void y_kernel(const _Float16* __restrict__ xh, const float* __restrict__ att,
              const float* __restrict__ wd, const float* __restrict__ bd,
              int C, float* __restrict__ Y /* (N,64,T,V) */) {
  int n = blockIdx.x / V_, w = blockIdx.x % V_;
  int KC = K_ * C;
  int KCp = (KC + 31) & ~31;
  int nch = KCp >> 5;                       // 1 (enc) or 6 (dec) K-chunks
  __shared__ __align__(32) _Float16 WF[4][6][32][16]; // [o-tile][chunk][lane][e]
  __shared__ __align__(32) _Float16 ZF[2][6][32][16]; // [t-subtile][chunk][lane][e]
  __shared__ float attW[K_][V_];
  __shared__ float bds[O_];
  int tid = threadIdx.x;

  for (int idx = tid; idx < KCp * O_; idx += 128) {
    int kc = idx / O_, o = idx % O_;
    _Float16 val = (_Float16)0.f;
    if (kc < KC) { int k = kc / C, c = kc % C; val = (_Float16)wd[(k * O_ + o) * C + c]; }
    int ot = o >> 4, nn = o & 15;
    int ch = kc >> 5, kl = kc & 31;
    WF[ot][ch][frag_lane(kl, nn)][frag_elem(kl)] = val;
  }
  for (int idx = tid; idx < K_ * V_; idx += 128) {
    int k = idx / V_, v = idx % V_;
    attW[k][v] = att[((size_t)(n * K_ + k) * V_ + v) * V_ + w];
  }
  if (tid < O_) bds[tid] = bd[tid] + bd[O_ + tid] + bd[2 * O_ + tid];
  __syncthreads();

  int lane = tid & 31, wv = tid >> 5;      // wave wv owns o-tile wv
  int h = lane >> 4, m = lane & 15;

  for (int t0 = 0; t0 < T_; t0 += 32) {
    for (int idx = tid; idx < 32 * KCp; idx += 128) {
      int tt = idx / KCp, kc = idx % KCp;
      float z = 0.f;
      int t = t0 + tt;
      if (kc < KC && t < T_) {
        int k = kc / C, c = kc % C;
        const _Float16* xr = &xh[((size_t)(n * C + c) * T_ + t) * V_];
        for (int v = 0; v < V_; ++v) z += (float)xr[v] * attW[k][v];
      }
      int ts = tt >> 4, mm = tt & 15;
      int ch = kc >> 5, kl = kc & 31;
      ZF[ts][ch][frag_lane(kl, mm)][frag_elem(kl)] = (_Float16)z;
    }
    __syncthreads();
#pragma unroll
    for (int ts = 0; ts < 2; ++ts) {
      v8f acc = {};
      for (int ch = 0; ch < nch; ++ch) {
        v16h a = *reinterpret_cast<const v16h*>(&ZF[ts][ch][lane][0]);
        v16h b = *reinterpret_cast<const v16h*>(&WF[wv][ch][lane][0]);
        acc = __builtin_amdgcn_wmma_f32_16x16x32_f16(false, a, false, b, (short)0,
                                                     acc, false, false);
      }
      int o = wv * 16 + m;
#pragma unroll
      for (int j = 0; j < 8; ++j) {
        int t = t0 + ts * 16 + j + 8 * h;
        if (t < T_)
          Y[((size_t)(n * O_ + o) * T_ + t) * V_ + w] = acc[j] + bds[o];
      }
    }
    __syncthreads();
  }
}

// ============================================================
// reductions for batch-norm statistics
// ============================================================
__device__ inline void block_reduce2(float& s, float& s2, float* sh) {
  int tid = threadIdx.x;
  sh[tid] = s; sh[256 + tid] = s2;
  __syncthreads();
  for (int st = 128; st > 0; st >>= 1) {
    if (tid < st) { sh[tid] += sh[tid + st]; sh[256 + tid] += sh[256 + tid + st]; }
    __syncthreads();
  }
  s = sh[0]; s2 = sh[256];
}

__global__ void stats_y_kernel(const float* __restrict__ Y,
                               float* __restrict__ ymean, float* __restrict__ yvar) {
  __shared__ float sh[512];
  int o = blockIdx.x;                       // 64 blocks
  const int per = T_ * V_;
  float s = 0.f, s2 = 0.f;
  for (int idx = threadIdx.x; idx < N_ * per; idx += 256) {
    int n = idx / per, r = idx % per;
    const float* p = &Y[((size_t)(n * O_ + o)) * per + r];
    __builtin_prefetch(p + 256, 0, 0);
    float y = *p;
    s += y; s2 += y * y;
  }
  block_reduce2(s, s2, sh);
  if (threadIdx.x == 0) {
    float cnt = (float)(N_ * per);
    float mn = s / cnt;
    ymean[o] = mn;
    yvar[o] = s2 / cnt - mn * mn;
  }
}

__global__ void stats_down_kernel(const _Float16* __restrict__ xh,
                                  const float* __restrict__ dw,
                                  const float* __restrict__ dbv,
                                  float* __restrict__ dmean, float* __restrict__ dvar) {
  __shared__ float sh[512];
  int o = blockIdx.x;
  const int per = T_ * V_;
  float w0 = dw[o * CIN_ + 0], w1 = dw[o * CIN_ + 1], w2 = dw[o * CIN_ + 2];
  float b = dbv[o];
  float s = 0.f, s2 = 0.f;
  for (int idx = threadIdx.x; idx < N_ * per; idx += 256) {
    int n = idx / per, r = idx % per;
    size_t base = ((size_t)n * CIN_) * per + r;
    float d = b + w0 * (float)xh[base] + w1 * (float)xh[base + per]
                + w2 * (float)xh[base + 2 * (size_t)per];
    s += d; s2 += d * d;
  }
  block_reduce2(s, s2, sh);
  if (threadIdx.x == 0) {
    float cnt = (float)(N_ * per);
    float mn = s / cnt;
    dmean[o] = mn;
    dvar[o] = s2 / cnt - mn * mn;
  }
}

// ============================================================
// K5: P = relu( BN(y) + down(x) ), stored f16 (N,64,T,V)
// ============================================================
__global__ void finalize_kernel(const float* __restrict__ Y,
                                const _Float16* __restrict__ xh, int C,
                                const float* __restrict__ gamma, const float* __restrict__ beta,
                                const float* __restrict__ ymean, const float* __restrict__ yvar,
                                int has_down,
                                const float* __restrict__ dw, const float* __restrict__ dbv,
                                const float* __restrict__ dgamma, const float* __restrict__ dbeta,
                                const float* __restrict__ dmean, const float* __restrict__ dvar,
                                _Float16* __restrict__ P) {
  const int per = T_ * V_;
  const long total = (long)N_ * O_ * per;
  for (long idx = (long)blockIdx.x * blockDim.x + threadIdx.x; idx < total;
       idx += (long)gridDim.x * blockDim.x) {
    int o = (int)((idx / per) % O_);
    int n = (int)(idx / ((long)O_ * per));
    int r = (int)(idx % per);
    float y = Y[idx];
    y = gamma[o] * (y - ymean[o]) * rsqrtf(yvar[o] + EPS_BN) + beta[o];
    float dn;
    if (has_down) {
      float d = dbv[o];
      for (int c = 0; c < CIN_; ++c)
        d += dw[o * CIN_ + c] * (float)xh[((size_t)(n * CIN_ + c)) * per + r];
      dn = dgamma[o] * (d - dmean[o]) * rsqrtf(dvar[o] + EPS_BN) + dbeta[o];
    } else {
      dn = (float)xh[idx];                 // identity: same (N,64,T,V) layout
    }
    y += dn;
    P[idx] = (_Float16)fmaxf(y, 0.f);
  }
}

// ============================================================
// K6: RAM_r[n,t,v,w] = sum_c p1[n,c,t,v]*p2[n,c,t,w]  (WMMA, K=64)
// Tiles staged via the Tensor Data Mover when the 5-arg builtin
// toolchain is detected (D#: 2D, data_size=2B, 25x64 tile,
// dim0_stride = T*V); otherwise cooperative vector loads.
// ============================================================
__global__ __launch_bounds__(128)
void ramr_kernel(const _Float16* __restrict__ p1, const _Float16* __restrict__ p2,
                 float* __restrict__ Rr /* (N,T,V,V) */) {
  int nt = blockIdx.x;
  int n = nt / T_, t = nt % T_;
  __shared__ __align__(16) _Float16 p1P[CENC_ * V_];   // packed 64 x 25
  __shared__ __align__(16) _Float16 p2P[CENC_ * V_];
  int tid = threadIdx.x;
  size_t tile_off = ((size_t)(n * CENC_) * T_ + t) * V_;

#if USE_TDM
  if (tid < 32) {   // wave 0 issues both TDM descriptors
    v8i_ g1;
    g1[0] = (int)(1u << 16);            // data_size = 1 (2 bytes); mask/flags 0
    g1[1] = (int)((unsigned)V_ << 16);  // tensor_dim0 = 25 (low 16)
    g1[2] = (int)((unsigned)CENC_ << 16); // tensor_dim1 = 64 (low 16)
    g1[3] = (int)((unsigned)V_ << 16);  // tile_dim0 = 25
    g1[4] = CENC_;                      // tile_dim1 = 64, tile_dim2 = 0
    g1[5] = T_ * V_;                    // tensor_dim0_stride = 7500 (elements)
    g1[6] = 0;
    g1[7] = 0;
    v4i_ gz = {0, 0, 0, 0};

    unsigned long long ga1 = (unsigned long long)(uintptr_t)(p1 + tile_off);
    v4u_ g0a;
    g0a[0] = 1u;                                    // count = 1, user descriptor
    g0a[1] = (unsigned)(uintptr_t)(void*)p1P;       // lds_addr
    g0a[2] = (unsigned)(ga1 & 0xffffffffu);
    g0a[3] = (unsigned)((ga1 >> 32) & 0x1ffffffu) | (2u << 30); // type=2
    __builtin_amdgcn_tensor_load_to_lds(g0a, g1, gz, gz, 0);

    unsigned long long ga2 = (unsigned long long)(uintptr_t)(p2 + tile_off);
    v4u_ g0b;
    g0b[0] = 1u;
    g0b[1] = (unsigned)(uintptr_t)(void*)p2P;
    g0b[2] = (unsigned)(ga2 & 0xffffffffu);
    g0b[3] = (unsigned)((ga2 >> 32) & 0x1ffffffu) | (2u << 30);
    __builtin_amdgcn_tensor_load_to_lds(g0b, g1, gz, gz, 0);

    __builtin_amdgcn_s_wait_tensorcnt(0);
  }
  __syncthreads();
#else
  for (int idx = tid; idx < CENC_ * V_; idx += 128) {
    p1P[idx] = p1[tile_off + (idx / V_) * (size_t)(T_ * V_) + (idx % V_)];
    p2P[idx] = p2[tile_off + (idx / V_) * (size_t)(T_ * V_) + (idx % V_)];
  }
  __syncthreads();
#endif

  int lane = tid & 31, wv = tid >> 5;
  int v0 = (wv & 1) * 16, w0 = (wv >> 1) * 16;
  int h = lane >> 4, m = lane & 15;
  int vA = v0 + m, vB = w0 + m;            // per-lane column into packed tile
  bool okA = vA < V_, okB = vB < V_;
  v8f acc = {};
#pragma unroll
  for (int c0 = 0; c0 < CENC_; c0 += 32) {
    v16h a, b;
#pragma unroll
    for (int e = 0; e < 16; ++e) {
      int kk = (e & 7) + 8 * h + 16 * (e >> 3);
      a[e] = okA ? p1P[(c0 + kk) * V_ + vA] : (_Float16)0.f;
      b[e] = okB ? p2P[(c0 + kk) * V_ + vB] : (_Float16)0.f;
    }
    acc = __builtin_amdgcn_wmma_f32_16x16x32_f16(false, a, false, b, (short)0,
                                                 acc, false, false);
  }
#pragma unroll
  for (int j = 0; j < 8; ++j) {
    int vv = v0 + j + 8 * h;
    int ww = w0 + m;
    if (vv < V_ && ww < V_)
      Rr[(size_t)nt * V_ * V_ + vv * V_ + ww] = acc[j];
  }
}

__global__ void stats_ramr_kernel(const float* __restrict__ Rr,
                                  float* __restrict__ rmean, float* __restrict__ rvar) {
  __shared__ float sh[512];
  int t = blockIdx.x;                      // 300 blocks
  float s = 0.f, s2 = 0.f;
  for (int idx = threadIdx.x; idx < N_ * V_ * V_; idx += 256) {
    int n = idx / (V_ * V_), r = idx % (V_ * V_);
    float x = Rr[((size_t)n * T_ + t) * V_ * V_ + r];
    s += x; s2 += x * x;
  }
  block_reduce2(s, s2, sh);
  if (threadIdx.x == 0) {
    float cnt = (float)(N_ * V_ * V_);
    float mn = s / cnt;
    rmean[t] = mn;
    rvar[t] = s2 / cnt - mn * mn;
  }
}

// ============================================================
// K7: final RAM composition, in-place on d_out
// ============================================================
__global__ void final_kernel(float* __restrict__ Rr,
                             const float* __restrict__ ramg,
                             const float* __restrict__ ab,
                             const float* __restrict__ rmean, const float* __restrict__ rvar,
                             const float* __restrict__ rgamma, const float* __restrict__ rbeta,
                             const float* __restrict__ ip) {
  int nt = blockIdx.x;
  int t = nt % T_;
  __shared__ float RL[V_][V_];
  __shared__ float da[V_], dbv[V_];
  int tid = threadIdx.x;
  float pct = (tanhf(ip[0]) + 1.f) * 0.5f;
  float mn = rmean[t], iv = rsqrtf(rvar[t] + EPS_BN);
  float g = rgamma[t], bt = rbeta[t];
  float mask = ab[nt];
  for (int idx = tid; idx < V_ * V_; idx += 128) {
    float r = Rr[(size_t)nt * V_ * V_ + idx];
    r = g * (r - mn) * iv + bt;
    r = (tanhf(r) + 1.f) * 0.5f;
    float ram = pct * r + (1.f - pct) * ramg[(size_t)nt * V_ * V_ + idx];
    ram *= mask;
    if (ram < 0.5f) ram = 0.f;
    RL[idx / V_][idx % V_] = ram;
  }
  __syncthreads();
  if (tid < V_) {                          // da[w] = sum over v
    float s = 0.f;
    for (int v = 0; v < V_; ++v) s += RL[v][tid];
    da[tid] = s;
  } else if (tid >= 32 && tid < 32 + V_) { // db[v] = sum over w
    int v = tid - 32;
    float s = 0.f;
    for (int w = 0; w < V_; ++w) s += RL[v][w];
    dbv[v] = s;
  }
  __syncthreads();
  for (int idx = tid; idx < V_ * V_; idx += 128) {
    int v = idx / V_, w = idx % V_;
    float norm = dbv[v] * da[w];
    Rr[(size_t)nt * V_ * V_ + idx] = RL[v][w] * rsqrtf(norm + 1e-5f);
  }
}

// ============================================================
// host-side orchestration
// ============================================================
static void unit_gcn(hipStream_t stream, const _Float16* xh, int C,
                     const float* A, const float* PA,
                     const float* wa, const float* ba,
                     const float* wb, const float* bb,
                     const float* wd, const float* bd,
                     const float* gamma, const float* beta,
                     const float* dw, const float* dbv,
                     const float* dgamma, const float* dbeta,   // null => identity down
                     float* att, float* Y, float* stats, _Float16* P) {
  att_kernel<<<N_ * K_, 128, 0, stream>>>(xh, wa, ba, wb, bb, C, att);
  softmax_kernel<<<N_ * K_, 32, 0, stream>>>(att, A, PA);
  y_kernel<<<N_ * V_, 128, 0, stream>>>(xh, att, wd, bd, C, Y);
  float* ymean = stats + 0;
  float* yvar  = stats + 64;
  float* dmean = stats + 128;
  float* dvar  = stats + 192;
  stats_y_kernel<<<O_, 256, 0, stream>>>(Y, ymean, yvar);
  if (dw) stats_down_kernel<<<O_, 256, 0, stream>>>(xh, dw, dbv, dmean, dvar);
  finalize_kernel<<<1024, 256, 0, stream>>>(Y, xh, C, gamma, beta, ymean, yvar,
                                            dw ? 1 : 0, dw, dbv, dgamma, dbeta,
                                            dmean, dvar, P);
}

extern "C" void kernel_launch(void* const* d_in, const int* in_sizes, int n_in,
                              void* d_out, int out_size, void* d_ws, size_t ws_size,
                              hipStream_t stream) {
  (void)in_sizes; (void)n_in; (void)out_size; (void)ws_size;
  const float* x        = (const float*)d_in[0];
  const float* A        = (const float*)d_in[1];
  const float* ip       = (const float*)d_in[2];
  const float* rgamma   = (const float*)d_in[3];
  const float* rbeta    = (const float*)d_in[4];
  const float* enc_PA   = (const float*)d_in[5];
  const float* enc_wa   = (const float*)d_in[6];
  const float* enc_ba   = (const float*)d_in[7];
  const float* enc_wb   = (const float*)d_in[8];
  const float* enc_bb   = (const float*)d_in[9];
  const float* enc_wd   = (const float*)d_in[10];
  const float* enc_bd   = (const float*)d_in[11];
  const float* enc_g    = (const float*)d_in[12];
  const float* enc_b    = (const float*)d_in[13];
  const float* enc_dw   = (const float*)d_in[14];
  const float* enc_db   = (const float*)d_in[15];
  const float* enc_dg   = (const float*)d_in[16];
  const float* enc_dbt  = (const float*)d_in[17];
  const float* dec_PA   = (const float*)d_in[18];
  const float* dec_wa   = (const float*)d_in[19];
  const float* dec_ba   = (const float*)d_in[20];
  const float* dec_wb   = (const float*)d_in[21];
  const float* dec_bb   = (const float*)d_in[22];
  const float* dec_wd   = (const float*)d_in[23];
  const float* dec_bd   = (const float*)d_in[24];
  const float* dec_g    = (const float*)d_in[25];
  const float* dec_b    = (const float*)d_in[26];

  // ---- workspace carve-up (256B aligned) ----
  char* base = (char*)d_ws;
  size_t off = 0;
  auto alloc = [&](size_t bytes) -> char* {
    off = (off + 255) & ~(size_t)255;
    char* p = base + off;
    off += bytes;
    return p;
  };
  const size_t xhalfs = (size_t)N_ * CIN_ * T_ * V_;     // 1.44M
  const size_t phalfs = (size_t)N_ * CENC_ * T_ * V_;    // 30.72M
  _Float16* xa_h = (_Float16*)alloc(xhalfs * 2);
  _Float16* xb_h = (_Float16*)alloc(xhalfs * 2);
  _Float16* P0   = (_Float16*)alloc(phalfs * 2);
  _Float16* P1   = (_Float16*)alloc(phalfs * 2);
  _Float16* P2   = (_Float16*)alloc(phalfs * 2);
  float* Y    = (float*)alloc((size_t)N_ * O_ * T_ * V_ * 4);
  float* att  = (float*)alloc((size_t)N_ * K_ * V_ * V_ * 4);
  float* ramg = (float*)alloc((size_t)N_ * T_ * V_ * V_ * 4);
  float* ab   = (float*)alloc((size_t)N_ * T_ * 4);
  float* stats = (float*)alloc(1024 * 4);
  float* rmean = stats + 256;
  float* rvar  = stats + 576;

  float* Rr = (float*)d_out;   // (N,T,V,V): RAM_r scratch, finished in place

  // ---- pipeline ----
  split_kernel<<<1024, 256, 0, stream>>>(x, xa_h, xb_h);
  ramg_kernel<<<N_ * T_, 128, 0, stream>>>(x, ramg, ab);

  // p1 = dec(enc(xa)); p2 = dec(enc(xb))
  unit_gcn(stream, xa_h, CIN_, A, enc_PA, enc_wa, enc_ba, enc_wb, enc_bb,
           enc_wd, enc_bd, enc_g, enc_b, enc_dw, enc_db, enc_dg, enc_dbt,
           att, Y, stats, P0);
  unit_gcn(stream, P0, CENC_, A, dec_PA, dec_wa, dec_ba, dec_wb, dec_bb,
           dec_wd, dec_bd, dec_g, dec_b, nullptr, nullptr, nullptr, nullptr,
           att, Y, stats, P1);
  unit_gcn(stream, xb_h, CIN_, A, enc_PA, enc_wa, enc_ba, enc_wb, enc_bb,
           enc_wd, enc_bd, enc_g, enc_b, enc_dw, enc_db, enc_dg, enc_dbt,
           att, Y, stats, P2);
  unit_gcn(stream, P2, CENC_, A, dec_PA, dec_wa, dec_ba, dec_wb, dec_bb,
           dec_wd, dec_bd, dec_g, dec_b, nullptr, nullptr, nullptr, nullptr,
           att, Y, stats, P0);

  ramr_kernel<<<N_ * T_, 128, 0, stream>>>(P1, P0, Rr);
  stats_ramr_kernel<<<T_, 256, 0, stream>>>(Rr, rmean, rvar);
  final_kernel<<<N_ * T_, 128, 0, stream>>>(Rr, ramg, ab, rmean, rvar,
                                            rgamma, rbeta, ip);
}
